// MultiHeadAttention_15290083574140
// MI455X (gfx1250) — matrix operand order repaired and checked by reading, not compiled, not run on hardware
//
#include <hip/hip_runtime.h>
#include <math.h>

// ---- problem shape -------------------------------------------------------
#define BSZ   8
#define CCH   4096
#define TDIM  512
#define NH    8
#define HDIM  64
#define OUTD  512
#define INV_SCALE 0.04419417382415922f   // 1/sqrt(512)

#define KCH    256              // halves per LDS-staged K chunk
#define LDSW_P (KCH + 8)        // padded LDS row stride (halves): rows 4 banks apart
#define LDSW_A (HDIM + 8)
#define KSPLIT 8                // split-K factor for the scores reduction

typedef __attribute__((ext_vector_type(16))) _Float16 v16h;
typedef __attribute__((ext_vector_type(8)))  _Float16 v8h;
typedef __attribute__((ext_vector_type(8)))  float    v8f;
typedef __attribute__((ext_vector_type(4)))  unsigned int u32x4;
typedef __attribute__((ext_vector_type(8)))  int      i32x8;
typedef __attribute__((ext_vector_type(4)))  int      i32x4;

#if defined(__has_builtin)
#  if __has_builtin(__builtin_amdgcn_tensor_load_to_lds)
#    define USE_TDM 1
#  endif
#endif
#ifndef USE_TDM
#  define USE_TDM 0
#endif

// ---- CDNA5 WMMA fragment index helpers (ISA 7.12.2, wave32) --------------
// A (16x32 f16): lane l -> row M=l&15 ; halfword i: K = i + 16*(l>=16)  (contiguous, matches B)
__device__ __forceinline__ int a_k(int lane, int i) {
  return i + ((lane >> 4) << 4);
}
// B (32x16 f16): lane l -> col N=l&15 ; halfword i: K = i + 16*(l>=16)  (contiguous order used by this code, same as A)
// C/D (16x16 f32): lane l -> N=l&15 ; VGPR r -> M = r + 8*(l>=16)

__device__ __forceinline__ v8f wmma16(v16h a, v16h b, v8f c) {
  return __builtin_amdgcn_wmma_f32_16x16x32_f16(false, a, false, b, (short)0, c,
                                                false, false);
}

// B fragment: 32 contiguous bytes from LDS (two ds_load_b128)
__device__ __forceinline__ v16h ld_b_lds(const _Float16* p) {
  v8h lo = *(const v8h*)p;
  v8h hi = *(const v8h*)(p + 8);
  v16h r;
#pragma unroll
  for (int i = 0; i < 8; ++i) { r[i] = lo[i]; r[i + 8] = hi[i]; }
  return r;
}

// ---- Tensor Data Mover: 2D f16 tile -> LDS with row padding --------------
// D# per ISA 8.3-8.5: group0 = {flags, lds_addr, global_addr, type=2};
// group1 = {data_size=2B, pad_enable, pad codes, tensor dims, tile dims, stride}.
#if USE_TDM
__device__ __forceinline__ void tdm_load_2d(const _Float16* g, unsigned lds_byte,
                                            int rows, int cols_halves,
                                            int stride_halves,
                                            int pad_int_code, int pad_amt_code) {
  unsigned long long ga = (unsigned long long)(size_t)g;
  u32x4 g0;
  g0[0] = 1u;                                    // count=1 (valid user D#)
  g0[1] = lds_byte;                              // lds_addr
  g0[2] = (unsigned)(ga & 0xffffffffu);          // global_addr[31:0]
  g0[3] = (unsigned)((ga >> 32) & 0x1ffffffu) | (2u << 30);  // addr[56:32] | type=2
  const int d0 = 1 << 20, d1 = 1 << 20;          // interior tile: never clip
  i32x8 g1;
  g1[0] = (1 << 16) | (1 << 20) | (pad_int_code << 22) | (pad_amt_code << 25);
  g1[1] = (d0 & 0xffff) << 16;                   // tensor_dim0 lo16
  g1[2] = ((d0 >> 16) & 0xffff) | ((d1 & 0xffff) << 16);
  g1[3] = ((d1 >> 16) & 0xffff) | (cols_halves << 16);   // tile_dim0
  g1[4] = rows & 0xffff;                         // tile_dim1 (tile_dim2=0)
  g1[5] = stride_halves;                         // tensor_dim0_stride lo32
  g1[6] = 0;
  g1[7] = 0;
  i32x4 z4 = {0, 0, 0, 0};
  i32x8 z8 = {0, 0, 0, 0, 0, 0, 0, 0};
  // 6-arg form on this toolchain: (g0, g1, g2, g3, extra, cpol)
  __builtin_amdgcn_tensor_load_to_lds(g0, g1, z4, z4, z8, 0);
}
#endif

__device__ __forceinline__ void stage_fallback(const _Float16* g, _Float16* s,
                                               int rows, int cols_halves,
                                               int gstride, int lstride,
                                               int tid, int nthr) {
  const int chunks = cols_halves >> 3;           // 16B chunks per row
  for (int idx = tid; idx < rows * chunks; idx += nthr) {
    int r = idx / chunks, c = (idx % chunks) << 3;
    *(v8h*)(s + r * lstride + c) = *(const v8h*)(g + (size_t)r * gstride + c);
  }
}

// stage rows x cols_halves f16 tile into padded LDS (row stride lstride halves)
__device__ __forceinline__ void stage_tile(const _Float16* g, _Float16* s,
                                           int rows, int cols_halves, int gstride,
                                           int lstride, int pad_int_code) {
#if USE_TDM
  if (threadIdx.x < 32) {
    tdm_load_2d(g, (unsigned)(size_t)(void*)s, rows, cols_halves, gstride,
                pad_int_code, /*pad 4 dwords=16B*/ 3);
    __builtin_amdgcn_s_wait_tensorcnt(0);
  }
#else
  (void)pad_int_code;
  stage_fallback(g, s, rows, cols_halves, gstride, lstride, threadIdx.x,
                 blockDim.x);
#endif
}

// ==========================================================================
// Kernel 0a/0b: weight fp32 -> f16 convert ; scratch zero
// ==========================================================================
__global__ void cvt_kernel(const float* __restrict__ s, _Float16* __restrict__ d,
                           int n) {
  int i = blockIdx.x * blockDim.x + threadIdx.x;
  if (i < n) d[i] = (_Float16)s[i];
}
__global__ void zero_kernel(float* __restrict__ p, int n) {
  int i = blockIdx.x * blockDim.x + threadIdx.x;
  if (i < n) p[i] = 0.f;
}

// ==========================================================================
// Kernel 1: Y[b,h,c,d] = sum_t X[b,c,t] * W[h,d,t]   (W pre-converted f16)
// grid (CCH/128, BSZ*NH), block 256; W tile staged in LDS via TDM.
// ==========================================================================
__global__ void proj_kernel(const float* __restrict__ X,
                            const _Float16* __restrict__ W16,
                            _Float16* __restrict__ Y) {
  __shared__ _Float16 sW[HDIM * LDSW_P];
  const int wave = threadIdx.x >> 5, lane = threadIdx.x & 31;
  const int mt = blockIdx.x * 8 + wave;
  const int bh = blockIdx.y, b = bh >> 3, h = bh & 7;
  const int m0 = mt * 16, row = lane & 15, khalf = (lane >> 4) << 4;

  const float*    Xb = X + (size_t)b * CCH * TDIM;
  const _Float16* Wh = W16 + (size_t)h * HDIM * TDIM;

  v8f acc[4] = {};
  for (int kc = 0; kc < TDIM; kc += KCH) {
    __syncthreads();                              // chunk consumed by all waves
    stage_tile(Wh + kc, sW, HDIM, KCH, TDIM, LDSW_P, /*64 qwords*/ 6);
    __syncthreads();
    for (int kk = 0; kk < KCH; kk += 32) {
      v16h a;
#pragma unroll
      for (int i = 0; i < 16; ++i)
        a[i] = (_Float16)Xb[(size_t)(m0 + row) * TDIM + kc + kk + a_k(lane, i)];
#pragma unroll
      for (int nt = 0; nt < 4; ++nt) {
        v16h bf = ld_b_lds(&sW[(nt * 16 + row) * LDSW_P + kk + khalf]);
        acc[nt] = wmma16(a, bf, acc[nt]);
      }
    }
  }
  _Float16* Yt = Y + ((size_t)b * NH + h) * CCH * HDIM;
  const int mb = (lane >> 4) << 3;
#pragma unroll
  for (int nt = 0; nt < 4; ++nt)
#pragma unroll
    for (int r = 0; r < 8; ++r)
      Yt[(size_t)(m0 + mb + r) * HDIM + nt * 16 + row] = (_Float16)acc[nt][r];
}

// ==========================================================================
// Kernel 2: partial scores, split-K over C; atomic f32 accumulate.
// grid (BSZ*NH, KSPLIT), block 128 (4 waves; wave -> 16(d) x 64(e) strip)
// ==========================================================================
__global__ void scores_kernel(const _Float16* __restrict__ qh,
                              const _Float16* __restrict__ kh,
                              float* __restrict__ S) {
  const int wave = threadIdx.x >> 5, lane = threadIdx.x & 31;
  const int bh = blockIdx.x, row = lane & 15, d0 = wave * 16;
  const int c_beg = blockIdx.y * (CCH / KSPLIT);

  const _Float16* Q = qh + (size_t)bh * CCH * HDIM;
  const _Float16* K = kh + (size_t)bh * CCH * HDIM;

  v8f acc[4] = {};
  for (int c0 = c_beg; c0 < c_beg + CCH / KSPLIT; c0 += 32) {
    v16h a;
#pragma unroll
    for (int i = 0; i < 16; ++i)               // A[m=d][k=c] = qh[c][d]
      a[i] = Q[(size_t)(c0 + a_k(lane, i)) * HDIM + d0 + row];
#pragma unroll
    for (int nt = 0; nt < 4; ++nt) {
      v16h bf;
#pragma unroll
      for (int i = 0; i < 16; ++i)             // B[k=c][n=e] = kh[c][e]
        bf[i] = K[(size_t)(c0 + i + ((lane >> 4) << 4)) * HDIM + nt * 16 + row];
      acc[nt] = wmma16(a, bf, acc[nt]);
    }
  }
  float* Sp = S + (size_t)bh * HDIM * HDIM;
  const int mb = (lane >> 4) << 3;
#pragma unroll
  for (int nt = 0; nt < 4; ++nt)
#pragma unroll
    for (int r = 0; r < 8; ++r)
      atomicAdd(&Sp[(size_t)(d0 + mb + r) * HDIM + nt * 16 + row], acc[nt][r]);
}

// ==========================================================================
// Kernel 3: softmax over rows of 64 (applies 1/SCALE), emits f16 probs.
// ==========================================================================
__global__ void softmax_kernel(const float* __restrict__ S,
                               _Float16* __restrict__ P) {
  const int r = blockIdx.x * blockDim.x + threadIdx.x;
  const float* p = S + (size_t)r * HDIM;
  _Float16*    o = P + (size_t)r * HDIM;
  float mx = -INFINITY;
  for (int e = 0; e < HDIM; ++e) mx = fmaxf(mx, p[e] * INV_SCALE);
  float sum = 0.f;
  for (int e = 0; e < HDIM; ++e) {
    float v = __expf(p[e] * INV_SCALE - mx);
    o[e] = (_Float16)v;
    sum += v;
  }
  float inv = 1.f / sum;
  for (int e = 0; e < HDIM; ++e) o[e] = (_Float16)((float)o[e] * inv);
}

// ==========================================================================
// Kernel 4: attn[c,d] = sum_e w[d,e]*vh[c,e] -> cat[b,c,h*64+d] (f16)
// grid (CCH/128, BSZ*NH), block 256; w tile (64x64 f16) staged via TDM.
// ==========================================================================
__global__ void attn_kernel(const _Float16* __restrict__ vh,
                            const _Float16* __restrict__ P,
                            _Float16* __restrict__ cat) {
  __shared__ _Float16 sW[HDIM * LDSW_A];
  const int wave = threadIdx.x >> 5, lane = threadIdx.x & 31;
  const int mt = blockIdx.x * 8 + wave;
  const int bh = blockIdx.y, b = bh >> 3, h = bh & 7;
  const int m0 = mt * 16, row = lane & 15, khalf = (lane >> 4) << 4;

  const _Float16* V = vh + (size_t)bh * CCH * HDIM;
  stage_tile(P + (size_t)bh * HDIM * HDIM, sW, HDIM, HDIM, HDIM, LDSW_A,
             /*16 qwords*/ 4);
  __syncthreads();

  v8f acc[4] = {};
#pragma unroll
  for (int e0 = 0; e0 < HDIM; e0 += 32) {
    v16h a;
#pragma unroll
    for (int i = 0; i < 16; ++i)               // A[m=c][k=e] = vh[c][e]
      a[i] = V[(size_t)(m0 + row) * HDIM + e0 + a_k(lane, i)];
#pragma unroll
    for (int nt = 0; nt < 4; ++nt) {           // B[k=e][n=d] = w[d][e]
      v16h bf = ld_b_lds(&sW[(nt * 16 + row) * LDSW_A + e0 + khalf]);
      acc[nt] = wmma16(a, bf, acc[nt]);
    }
  }
  _Float16* Cb = cat + (size_t)b * CCH * OUTD + (size_t)h * HDIM;
  const int mb = (lane >> 4) << 3;
#pragma unroll
  for (int nt = 0; nt < 4; ++nt)
#pragma unroll
    for (int r = 0; r < 8; ++r)
      Cb[(size_t)(m0 + mb + r) * OUTD + nt * 16 + row] = (_Float16)acc[nt][r];
}

// ==========================================================================
// Kernel 5: out[r,n] = sum_k cat[r,k] * Wo[n,k]; Wo slice staged via TDM.
// grid (BSZ*CCH/128, OUTD/64), block 256.
// ==========================================================================
__global__ void out_kernel(const _Float16* __restrict__ cat,
                           const _Float16* __restrict__ Wo16,
                           float* __restrict__ out) {
  __shared__ _Float16 sW[HDIM * LDSW_P];
  const int wave = threadIdx.x >> 5, lane = threadIdx.x & 31;
  const int mt = blockIdx.x * 8 + wave;
  const int nb = blockIdx.y * 64;
  const int m0 = mt * 16, row = lane & 15, khalf = (lane >> 4) << 4;

  const _Float16* Wb = Wo16 + (size_t)nb * OUTD;

  v8f acc[4] = {};
  for (int kc = 0; kc < OUTD; kc += KCH) {
    __syncthreads();
    stage_tile(Wb + kc, sW, 64, KCH, OUTD, LDSW_P, 6);
    __syncthreads();
    for (int kk = 0; kk < KCH; kk += 32) {
      v16h a;
#pragma unroll
      for (int i = 0; i < 16; ++i)
        a[i] = cat[(size_t)(m0 + row) * OUTD + kc + kk + a_k(lane, i)];
#pragma unroll
      for (int nt = 0; nt < 4; ++nt) {
        v16h bf = ld_b_lds(&sW[(nt * 16 + row) * LDSW_P + kk + khalf]);
        acc[nt] = wmma16(a, bf, acc[nt]);
      }
    }
  }
  const int mb = (lane >> 4) << 3;
#pragma unroll
  for (int nt = 0; nt < 4; ++nt)
#pragma unroll
    for (int r = 0; r < 8; ++r)
      out[(size_t)(m0 + mb + r) * OUTD + nb + nt * 16 + row] = acc[nt][r];
}

// ==========================================================================
extern "C" void kernel_launch(void* const* d_in, const int* in_sizes, int n_in,
                              void* d_out, int out_size, void* d_ws, size_t ws_size,
                              hipStream_t stream) {
  const float* q  = (const float*)d_in[0];
  const float* k  = (const float*)d_in[1];
  const float* v  = (const float*)d_in[2];
  const float* Wq = (const float*)d_in[3];
  const float* Wk = (const float*)d_in[4];
  const float* Wv = (const float*)d_in[5];
  const float* Wo = (const float*)d_in[6];
  float* out = (float*)d_out;

  const size_t QKV = (size_t)BSZ * NH * CCH * HDIM * 2;   // 32 MiB each
  const size_t SCR = (size_t)BSZ * NH * HDIM * HDIM * 4;  // 1 MiB
  const size_t WSM = (size_t)BSZ * NH * HDIM * HDIM * 2;  // 512 KiB
  const size_t CAT = (size_t)BSZ * CCH * OUTD * 2;        // 32 MiB
  const size_t W16 = (size_t)NH * HDIM * TDIM * 2;        // 512 KiB (== Wo f16)

  char* ws = (char*)d_ws;
  _Float16* qh   = (_Float16*)(ws);
  _Float16* kh   = (_Float16*)(ws + QKV);
  _Float16* vh   = (_Float16*)(ws + 2 * QKV);
  float*    scr  = (float*)(ws + 3 * QKV);
  _Float16* wsm  = (_Float16*)(ws + 3 * QKV + SCR);
  _Float16* cat  = (_Float16*)(ws + 3 * QKV + SCR + WSM);
  _Float16* wq16 = (_Float16*)(ws + 3 * QKV + SCR + WSM + CAT);
  _Float16* wk16 = wq16 + (W16 / 2);
  _Float16* wv16 = wk16 + (W16 / 2);
  _Float16* wo16 = wv16 + (W16 / 2);

  const int NW = NH * HDIM * TDIM;                        // 262144 (== Wo elems)
  cvt_kernel<<<NW / 256, 256, 0, stream>>>(Wq, wq16, NW);
  cvt_kernel<<<NW / 256, 256, 0, stream>>>(Wk, wk16, NW);
  cvt_kernel<<<NW / 256, 256, 0, stream>>>(Wv, wv16, NW);
  cvt_kernel<<<NW / 256, 256, 0, stream>>>(Wo, wo16, NW);
  zero_kernel<<<(BSZ * NH * HDIM * HDIM) / 256, 256, 0, stream>>>(
      scr, BSZ * NH * HDIM * HDIM);

  const dim3 bigGrid(CCH / 128, BSZ * NH);
  proj_kernel<<<bigGrid, 256, 0, stream>>>(q, wq16, qh);
  proj_kernel<<<bigGrid, 256, 0, stream>>>(k, wk16, kh);
  proj_kernel<<<bigGrid, 256, 0, stream>>>(v, wv16, vh);

  scores_kernel<<<dim3(BSZ * NH, KSPLIT), 128, 0, stream>>>(qh, kh, scr);
  softmax_kernel<<<(BSZ * NH * HDIM) / 256, 256, 0, stream>>>(scr, wsm);

  attn_kernel<<<bigGrid, 256, 0, stream>>>(vh, wsm, cat);
  out_kernel<<<dim3((BSZ * CCH) / 128, OUTD / 64), 256, 0, stream>>>(cat, wo16, out);
}
